// WinogradConv2d_54262616818114
// MI455X (gfx1250) — matrix operand order repaired and checked by reading, not compiled, not run on hardware
//
#include <hip/hip_runtime.h>
#include <hip/hip_bf16.h>

// ---------------------------------------------------------------------------
// Winograd F(2x2,3x3) conv for gfx1250 (MI455X, wave32, WMMA).
//   B=16, Cin=128, Cout=256, H=W=64  -> 16384 2x2 tiles.
// 3-stage pipeline: filter xform (U, f16) -> input xform (V, f16) ->
// fused 16x GEMM (v_wmma_f32_16x16x32_f16, f32 accum) + output xform.
// Workspace: V = 64 MB @ off 0, U = 1 MB @ off 64 MB.
// ---------------------------------------------------------------------------

typedef _Float16 half_t;
typedef half_t v16h __attribute__((ext_vector_type(16)));
typedef half_t v8h  __attribute__((ext_vector_type(8)));
typedef half_t v2h  __attribute__((ext_vector_type(2)));
typedef float  v8f  __attribute__((ext_vector_type(8)));
typedef float  v4f  __attribute__((ext_vector_type(4)));
typedef float  v2f  __attribute__((ext_vector_type(2)));

#define NBATCH 16
#define CIN    128
#define COUT   256
#define HW     64
#define NTILES 16384                 // 16 * 32 * 32
#define VPLANE ((size_t)NTILES*CIN)  // halves per pq plane of V
#define UPLANE ((size_t)COUT*CIN)    // halves per pq plane of U

// ---------------------------------------------------------------------------
// Kernel 1: U[pq][k][c] = (G w G^T), f16.  One thread per (k,c).
// ---------------------------------------------------------------------------
__global__ __launch_bounds__(256) void wino_filter_xform(
    const float* __restrict__ wgt, half_t* __restrict__ U) {
  int g = blockIdx.x * 256 + threadIdx.x;          // 32768 threads
  int k = g >> 7, c = g & 127;
  const float* wp = wgt + (size_t)(k * 128 + c) * 9;
  float gm[3][3];
#pragma unroll
  for (int y = 0; y < 3; ++y)
#pragma unroll
    for (int x = 0; x < 3; ++x) gm[y][x] = wp[y * 3 + x];
  // rows: a = G * g   (G = [[1,0,0],[.5,.5,.5],[.5,-.5,.5],[0,0,1]])
  float a[4][3];
#pragma unroll
  for (int x = 0; x < 3; ++x) {
    a[0][x] = gm[0][x];
    a[1][x] = 0.5f * (gm[0][x] + gm[1][x] + gm[2][x]);
    a[2][x] = 0.5f * (gm[0][x] - gm[1][x] + gm[2][x]);
    a[3][x] = gm[2][x];
  }
  size_t base = (size_t)k * 128 + c;
#pragma unroll
  for (int p = 0; p < 4; ++p) {
    float u0 = a[p][0];
    float u1 = 0.5f * (a[p][0] + a[p][1] + a[p][2]);
    float u2 = 0.5f * (a[p][0] - a[p][1] + a[p][2]);
    float u3 = a[p][2];
    U[(size_t)(p * 4 + 0) * UPLANE + base] = (half_t)u0;
    U[(size_t)(p * 4 + 1) * UPLANE + base] = (half_t)u1;
    U[(size_t)(p * 4 + 2) * UPLANE + base] = (half_t)u2;
    U[(size_t)(p * 4 + 3) * UPLANE + base] = (half_t)u3;
  }
}

// BT d BT^T helper (BT = [[1,0,-1,0],[0,1,1,0],[0,-1,1,0],[0,1,0,-1]])
__device__ __forceinline__ void bt_xform(const float d[4][4], float v[4][4]) {
  float w[4][4];
#pragma unroll
  for (int y = 0; y < 4; ++y) {
    w[0][y] = d[0][y] - d[2][y];
    w[1][y] = d[1][y] + d[2][y];
    w[2][y] = d[2][y] - d[1][y];
    w[3][y] = d[1][y] - d[3][y];
  }
#pragma unroll
  for (int p = 0; p < 4; ++p) {
    v[p][0] = w[p][0] - w[p][2];
    v[p][1] = w[p][1] + w[p][2];
    v[p][2] = w[p][2] - w[p][1];
    v[p][3] = w[p][1] - w[p][3];
  }
}

// ---------------------------------------------------------------------------
// Kernel 2: V[pq][tile][c] f16. One thread per (tile, channel-pair).
// Tile-fastest mapping -> coalesced X reads (reads dominate: 128MB vs 64MB).
// ---------------------------------------------------------------------------
__global__ __launch_bounds__(256) void wino_input_xform(
    const float* __restrict__ X, half_t* __restrict__ V) {
  int g = blockIdx.x * 256 + threadIdx.x;          // 1,048,576 threads
  int tile = g & (NTILES - 1);
  int c = (g >> 14) << 1;                          // channel pair
  int b = tile >> 10, i = (tile >> 5) & 31, j = tile & 31;
  const float* x0 = X + ((size_t)b * CIN + c) * (HW * HW);
  const float* x1 = x0 + HW * HW;
  int r0 = i * 2 - 1, q0 = j * 2 - 1;
  float d0[4][4], d1[4][4];
#pragma unroll
  for (int y = 0; y < 4; ++y) {
    int r = r0 + y;
#pragma unroll
    for (int x = 0; x < 4; ++x) {
      int cc = q0 + x;
      bool ok = ((unsigned)r < 64u) && ((unsigned)cc < 64u);
      int idx = (r << 6) + cc;
      d0[y][x] = ok ? x0[idx] : 0.0f;
      d1[y][x] = ok ? x1[idx] : 0.0f;
    }
  }
  float v0[4][4], v1[4][4];
  bt_xform(d0, v0);
  bt_xform(d1, v1);
  half_t* Vo = V + (size_t)tile * CIN + c;
#pragma unroll
  for (int p = 0; p < 4; ++p)
#pragma unroll
    for (int q = 0; q < 4; ++q) {
      v2h pv = {(half_t)v0[p][q], (half_t)v1[p][q]};
      *(v2h*)(Vo + (size_t)(p * 4 + q) * VPLANE) = pv;
    }
}

// ---------------------------------------------------------------------------
// Kernel 3: fused GEMM (16 pq-planes accumulated per wave) + output xform.
// Wave tile: 16 tiles (M) x 16 couts (N); WG = 8 waves = 32 tiles x 64 couts.
// A (V) and B (U) loaded directly into wave32 WMMA register layouts:
//   A 16x32 f16: lane<16 -> K c0+0..7 / c0+16..23 ; lane>=16 -> +8 offset.
//   B 32x16 f16: lane<16 -> K c0+0..7 / c0+16..23 ; lane>=16 -> +8 offset (N = lane&15).
// ---------------------------------------------------------------------------
__global__ __launch_bounds__(256) void wino_gemm_out(
    const half_t* __restrict__ V, const half_t* __restrict__ U,
    const float* __restrict__ bias, float* __restrict__ out) {
  __shared__ float lds[8 * 1024];                  // 4KB per wave
  int lane = threadIdx.x & 31;
  int w    = threadIdx.x >> 5;
  int wm = w & 1, wn = w >> 1;
  int lmod = lane & 15, lhi = lane >> 4;
  int tb = blockIdx.x * 32 + wm * 16;              // first tile of wave block
  int nb = blockIdx.y * 64 + wn * 16;              // first cout of wave block

  const half_t* aBase = V + ((size_t)(tb + lmod)) * CIN + lhi * 8;
  const half_t* bBase = U + ((size_t)(nb + lmod)) * CIN + lhi * 8;

  v8f acc[16];
#pragma unroll
  for (int pq = 0; pq < 16; ++pq) acc[pq] = (v8f){0.f,0.f,0.f,0.f,0.f,0.f,0.f,0.f};

  for (int c0 = 0; c0 < CIN; c0 += 32) {
#pragma unroll
    for (int pq = 0; pq < 16; ++pq) {
      const v8h* pa = (const v8h*)(aBase + (size_t)pq * VPLANE + c0);
      v8h alo = pa[0];                 // K c0+{0..7 | 8..15}
      v8h ahi = pa[2];                 // K c0+16+{0..7 | 8..15}
      const v8h* pb = (const v8h*)(bBase + (size_t)pq * UPLANE + c0);
      v8h blo = pb[0];                 // K c0+{0..7 | 8..15}
      v8h bhi = pb[2];                 // K c0+16+{0..7 | 8..15}
      v16h a = __builtin_shufflevector(alo, ahi, 0,1,2,3,4,5,6,7,8,9,10,11,12,13,14,15);
      v16h bm = __builtin_shufflevector(blo, bhi, 0,1,2,3,4,5,6,7,8,9,10,11,12,13,14,15);
      acc[pq] = __builtin_amdgcn_wmma_f32_16x16x32_f16(
          false, a, false, bm, (short)0, acc[pq], false, false);
    }
  }

  // Output transform: each (lane, acc element r) owns the full 4x4 M for one
  // (tile m = r + 8*lhi, cout n = lmod) pair -> purely lane-local AT*M*AT^T.
  float bv = bias[nb + lmod];
  float* myLds = lds + w * 1024;
#pragma unroll
  for (int r = 0; r < 8; ++r) {
    int m = r + lhi * 8;
    float M[16];
#pragma unroll
    for (int pq = 0; pq < 16; ++pq) M[pq] = acc[pq][r];
    float t00 = M[0] + M[4] + M[8],  t01 = M[1] + M[5] + M[9];
    float t02 = M[2] + M[6] + M[10], t03 = M[3] + M[7] + M[11];
    float t10 = M[4] - M[8] - M[12], t11 = M[5] - M[9] - M[13];
    float t12 = M[6] - M[10] - M[14],t13 = M[7] - M[11] - M[15];
    v4f y = { t00 + t01 + t02 + bv, t01 - t02 - t03 + bv,
              t10 + t11 + t12 + bv, t11 - t12 - t13 + bv };
    int mp = m ^ (lmod & 7);                        // bank-spread swizzle
    *(v4f*)(myLds + (size_t)(lmod * 16 + mp) * 4) = y;
  }
  __syncthreads();

  // Coalesced writeback: per (cout n, row u) the 16 tiles x 2 cols form a
  // contiguous 128B row segment; lanes 0-15 cover u=0, lanes 16-31 cover u=1.
  int b = blockIdx.x >> 5;
  int i = blockIdx.x & 31;
#pragma unroll
  for (int n = 0; n < 16; ++n) {
    int mp = lmod ^ (n & 7);
    v2f val = *(v2f*)(myLds + (size_t)(n * 16 + mp) * 4 + lhi * 2);
    int k = nb + n;
    size_t off = (((size_t)b * COUT + k) * HW + (2 * i + lhi)) * HW
               + wm * 32 + 2 * lmod;
    *(v2f*)(out + off) = val;
  }
}

// ---------------------------------------------------------------------------
extern "C" void kernel_launch(void* const* d_in, const int* in_sizes, int n_in,
                              void* d_out, int out_size, void* d_ws, size_t ws_size,
                              hipStream_t stream) {
  (void)in_sizes; (void)n_in; (void)out_size; (void)ws_size;
  const float* X    = (const float*)d_in[0];
  const float* wgt  = (const float*)d_in[1];
  const float* bias = (const float*)d_in[2];
  float* out = (float*)d_out;

  half_t* Vw = (half_t*)d_ws;                                        // 64 MB
  half_t* Uw = (half_t*)((char*)d_ws + (size_t)16 * VPLANE * 2);     // +1 MB

  wino_filter_xform<<<128, 256, 0, stream>>>(wgt, Uw);
  wino_input_xform<<<4096, 256, 0, stream>>>(X, Vw);
  wino_gemm_out<<<dim3(512, 4), 256, 0, stream>>>(Vw, Uw, bias, out);
}
